// NetVLAD_13280038880041
// MI455X (gfx1250) — matrix operand order repaired and checked by reading, not compiled, not run on hardware
//
#include <hip/hip_runtime.h>
#include <cstdint>
#include <cstddef>

#define CC     128
#define KK     64
#define HH     40
#define WW     200
#define PP     8000   // HH*WW
#define NN     2
#define OUTD   256
#define MCHUNK 16     // maps per conv-stack pass (NN*KK = 128 total)

typedef __attribute__((ext_vector_type(16))) __bf16 v16bf;
typedef __attribute__((ext_vector_type(8)))  float  v8f;
typedef int vec4i __attribute__((vector_size(16)));

union Frag { unsigned int u[8]; v16bf v; };

#if __has_builtin(__builtin_amdgcn_global_load_async_to_lds_b128)
#define ASYNC_LDS 1
#else
#define ASYNC_LDS 0
#endif

__device__ __forceinline__ void cp16_lds(unsigned short* dst_lds, const unsigned short* src) {
#if ASYNC_LDS
  __builtin_amdgcn_global_load_async_to_lds_b128(
      (__attribute__((address_space(1))) vec4i*)src,
      (__attribute__((address_space(3))) vec4i*)dst_lds, 0, 0);
#else
  *(uint4*)dst_lds = *(const uint4*)src;
#endif
}

__device__ __forceinline__ void wait_async_lds() {
#if ASYNC_LDS
#if __has_builtin(__builtin_amdgcn_s_wait_asynccnt)
  __builtin_amdgcn_s_wait_asynccnt(0);
#else
  asm volatile("s_wait_asynccnt 0x0" ::: "memory");
#endif
#endif
}

__device__ __forceinline__ unsigned short f2bf(float f) {
  unsigned int u = __float_as_uint(f);
  unsigned int r = u + 0x7FFFu + ((u >> 16) & 1u);   // round-to-nearest-even
  return (unsigned short)(r >> 16);
}
__device__ __forceinline__ float bf2f(unsigned short h) {
  return __uint_as_float(((unsigned int)h) << 16);
}
__device__ __forceinline__ void atomicMaxF(float* a, float v) {
  if (v >= 0.0f) atomicMax((int*)a, __float_as_int(v));
  else           atomicMin((unsigned int*)a, __float_as_uint(v));
}

// ---------------------------------------------------------------- init
__global__ void k_init(float* maxbuf) {
  int i = blockIdx.x * blockDim.x + threadIdx.x;
  if (i < NN * KK * CC) maxbuf[i] = __uint_as_float(0xFF800000u); // -inf
}

// ------------------------------------------------- weight pack f32->bf16
// wpk layout: [tap 25][co 128][ci 128]; src: [co][ci][5][5]
__global__ void k_pack(const float* w, unsigned short* wpk) {
  int i = blockIdx.x * blockDim.x + threadIdx.x;
  if (i >= 25 * CC * CC) return;
  int tap = i >> 14;
  int r = i & 16383;
  int co = r >> 7, ci = r & 127;
  wpk[i] = f2bf(w[(size_t)(co * CC + ci) * 25 + tap]);
}

// ------------------------------------------- soft assignment + distances
// soft [N][K][P] f32 ; distb [N*K][P] bf16
__global__ __launch_bounds__(128)
void k_assign(const float* x, const float* cw, const float* cb,
              const float* cents, float* soft, unsigned short* distb) {
  extern __shared__ float sma[];
  float* xls = sma;            // [c 128][t 128]
  float* lg  = sma + CC * 128; // [t 128][65]
  int n = blockIdx.y, t = threadIdx.x;
  int p = blockIdx.x * 128 + t;
  bool ok = p < PP;
  int pc = ok ? p : (PP - 1);
  const float* xb = x + (size_t)n * CC * PP;
  for (int c = 0; c < CC; ++c) xls[c * 128 + t] = xb[(size_t)c * PP + pc];
  __syncthreads();
  float xsq = 0.f;
  for (int c = 0; c < CC; ++c) { float v = xls[c * 128 + t]; xsq += v * v; }
  float mx = -3.4e38f;
  for (int k = 0; k < KK; ++k) {
    float acc = cb[k];
    for (int c = 0; c < CC; ++c) acc += xls[c * 128 + t] * cw[k * CC + c];
    lg[t * 65 + k] = acc;
    mx = fmaxf(mx, acc);
  }
  float s = 0.f;
  for (int k = 0; k < KK; ++k) {
    float e = __expf(lg[t * 65 + k] - mx);
    lg[t * 65 + k] = e;
    s += e;
  }
  float inv = 1.f / s;
  for (int k = 0; k < KK; ++k) {
    if (ok) soft[((size_t)n * KK + k) * PP + p] = lg[t * 65 + k] * inv;
    float xc = 0.f, csq = 0.f;
    for (int c = 0; c < CC; ++c) {
      float cv = cents[k * CC + c];
      xc += xls[c * 128 + t] * cv;
      csq += cv * cv;
    }
    float d = sqrtf(fmaxf(xsq + csq - 2.f * xc, 0.f));
    if (ok) distb[((size_t)n * KK + k) * PP + p] = f2bf(d);
  }
}

// ------------------------------------------------------------- softsum
__global__ __launch_bounds__(256)
void k_softsum(const float* soft, float* softsum) {
  __shared__ float red[256];
  int b = blockIdx.x, t = threadIdx.x;
  const float* s = soft + (size_t)b * PP;
  float a = 0.f;
  for (int p = t; p < PP; p += 256) a += s[p];
  red[t] = a; __syncthreads();
  for (int st = 128; st > 0; st >>= 1) { if (t < st) red[t] += red[t + st]; __syncthreads(); }
  if (t == 0) softsum[b] = red[0];
}

// ------------------------------------------------------------------ VLAD
__global__ __launch_bounds__(128)
void k_vlad(const float* x, const float* soft, const float* softsum,
            const float* cents, float* vlad) {
  extern __shared__ float smv[];
  float* xs = smv;              // [c 128][p pad 129]
  float* sv = smv + 128 * 129;  // [128]
  int k = blockIdx.x, n = blockIdx.y, t = threadIdx.x;
  const float* xb = x + (size_t)n * CC * PP;
  const float* sb = soft + ((size_t)n * KK + k) * PP;
  float acc = 0.f;
  for (int p0 = 0; p0 < PP; p0 += 128) {
    int pcnt = min(PP - p0, 128);
    int pl = min(p0 + t, PP - 1);
    for (int c = 0; c < CC; ++c) xs[c * 129 + t] = xb[(size_t)c * PP + pl];
    sv[t] = (t < pcnt) ? sb[p0 + t] : 0.f;
    __syncthreads();
    for (int pi = 0; pi < pcnt; ++pi) acc += sv[pi] * xs[t * 129 + pi];
    __syncthreads();
  }
  vlad[((size_t)n * KK + k) * CC + t] = acc - softsum[n * KK + k] * cents[k * CC + t];
}

// ------------------------------------------- conv1 (1->128 ch) + bn + lrelu
// out layout: h1[ml][h][w][co] bf16
__global__ __launch_bounds__(256)
void k_conv1(const unsigned short* distb, const float* c1w, const float* c1b,
             const float* g, const float* bb, const float* mm, const float* vv,
             unsigned short* h1, int map_base) {
  __shared__ unsigned short in_s[5][204];
  __shared__ float w_s[128][25];
  int h = blockIdx.x, ml = blockIdx.y, t = threadIdx.x;
  int mg = map_base + ml;
  const unsigned short* db = distb + (size_t)mg * PP;
  for (int idx = t; idx < 5 * 204; idx += 256) {
    int row = idx / 204, col = idx % 204;
    int gh = h - 2 + row, gw = col - 2;
    unsigned short v = 0;
    if (gh >= 0 && gh < HH && gw >= 0 && gw < WW) v = db[gh * WW + gw];
    in_s[row][col] = v;
  }
  for (int idx = t; idx < 128 * 25; idx += 256) w_s[idx / 25][idx % 25] = c1w[idx];
  __syncthreads();
  for (int idx = t; idx < WW * CC; idx += 256) {
    int wcol = idx >> 7, co = idx & 127;
    float acc = 0.f;
    #pragma unroll
    for (int dy = 0; dy < 5; ++dy)
      #pragma unroll
      for (int dx = 0; dx < 5; ++dx)
        acc += bf2f(in_s[dy][wcol + dx]) * w_s[co][dy * 5 + dx];
    acc += c1b[co];
    float sc = g[co] * rsqrtf(vv[co] + 1e-5f);
    float y = acc * sc + (bb[co] - mm[co] * sc);
    y = (y > 0.f) ? y : 0.2f * y;
    h1[(((size_t)ml * HH + h) * WW + wcol) * CC + co] = f2bf(y);
  }
}

// --------------------- conv2/conv3 implicit GEMM via v_wmma_f32_16x16x32_bf16
// Block: 2 output rows x 64 px x all 128 co; 8 waves, each co-stripe of 16.
// mode 0: +bias, bn, lrelu -> hout (bf16, [ml][h][w][co])
// mode 1: +bias, fused global max-pool -> maxbuf (f32 atomic max)
__global__ __launch_bounds__(256)
void k_convg(const unsigned short* hin, const unsigned short* wpk,
             const float* bias, const float* g, const float* b2,
             const float* m2, const float* v2,
             unsigned short* hout, float* maxbuf, int mode, int map_base) {
  extern __shared__ __align__(16) unsigned short smem[];
  unsigned short* in_s = smem;                // [6][68][128] bf16
  unsigned short* w_s  = smem + 6 * 68 * 128; // [128][128]   bf16
  int t = threadIdx.x;
  int lane = t & 31, wave = t >> 5;
  int h0 = blockIdx.y * 2, ml = blockIdx.z;
  int w0 = blockIdx.x * 64;

  // stage input patch rows h0-2..h0+3, cols w0-2..w0+65, all ci (zero-padded)
  {
    const unsigned short* hb = hin + (size_t)ml * HH * WW * CC;
    for (int idx = t; idx < 6 * 68 * 16; idx += 256) {
      int cidx = idx & 15;
      int rc = idx >> 4;
      int col = rc % 68, row = rc / 68;
      int gh = h0 - 2 + row, gw = w0 - 2 + col;
      unsigned short* dst = in_s + ((size_t)rc << 7) + (cidx << 3);
      if (gh >= 0 && gh < HH && gw >= 0 && gw < WW)
        cp16_lds(dst, hb + (((size_t)gh * WW + gw) << 7) + (cidx << 3));
      else
        *(uint4*)dst = make_uint4(0u, 0u, 0u, 0u);
    }
  }

  v8f acc[2][4];
  #pragma unroll
  for (int r = 0; r < 2; ++r)
    #pragma unroll
    for (int j = 0; j < 4; ++j)
      #pragma unroll
      for (int v = 0; v < 8; ++v) acc[r][j][v] = 0.f;

  int nidx = lane & 15, half = lane >> 4;
  int co_row = wave * 16 + nidx;  // A-matrix row (co) for this lane
  const unsigned int* isu = (const unsigned int*)in_s;
  const unsigned int* wsu = (const unsigned int*)w_s;

  for (int tap = 0; tap < 25; ++tap) {
    __syncthreads();
    {
      const unsigned short* wsrc = wpk + (size_t)tap * CC * CC;
      for (int idx = t; idx < (CC * CC) / 8; idx += 256)
        cp16_lds(w_s + (idx << 3), wsrc + (idx << 3));
    }
    if (tap + 1 < 25) __builtin_prefetch(wpk + (size_t)(tap + 1) * CC * CC, 0, 1);
    wait_async_lds();
    __syncthreads();
    int dy = tap / 5, dx = tap % 5;
    #pragma unroll
    for (int kk = 0; kk < 4; ++kk) {
      int ci0 = kk * 32;
      // A fragment (16x32 bf16): lanes 0-15 M=0-15; dword v: K = (v<4 ? 2v : 16+2(v-4)) + half*8
      Frag a;
      #pragma unroll
      for (int v = 0; v < 8; ++v) {
        int kb = ((v < 4) ? 2 * v : 16 + 2 * (v - 4)) + half * 8;
        a.u[v] = wsu[(co_row * CC + ci0 + kb) >> 1];
      }
      #pragma unroll
      for (int r = 0; r < 2; ++r) {
        #pragma unroll
        for (int j = 0; j < 4; ++j) {
          int col = j * 16 + nidx + dx;
          // B fragment (32x16 bf16): dword v: K = (v<4 ? 2v : 16+2(v-4)) + half*8
          Frag bf;
          #pragma unroll
          for (int v = 0; v < 8; ++v) {
            int kb = ((v < 4) ? 2 * v : 16 + 2 * (v - 4)) + half * 8;
            bf.u[v] = isu[((((r + dy) * 68 + col) << 7) + ci0 + kb) >> 1];
          }
          acc[r][j] = __builtin_amdgcn_wmma_f32_16x16x32_bf16(
              false, a.v, false, bf.v, (short)0, acc[r][j], false, false);
        }
      }
    }
  }

  // C/D layout: lane -> n=lane&15 (px), VGPR v -> m = half*8+v (co)
  int cob = wave * 16 + half * 8;
  if (mode == 0) {
    float sc[8], sh[8], bs[8];
    #pragma unroll
    for (int v = 0; v < 8; ++v) {
      int c = cob + v;
      float s = g[c] * rsqrtf(v2[c] + 1e-5f);
      sc[v] = s; sh[v] = b2[c] - m2[c] * s; bs[v] = bias[c];
    }
    #pragma unroll
    for (int r = 0; r < 2; ++r) {
      int h = h0 + r;
      #pragma unroll
      for (int j = 0; j < 4; ++j) {
        int px = w0 + j * 16 + nidx;
        if (px < WW) {
          union { unsigned short o[8]; uint4 q; } pk;
          #pragma unroll
          for (int v = 0; v < 8; ++v) {
            float y = (acc[r][j][v] + bs[v]) * sc[v] + sh[v];
            y = (y > 0.f) ? y : 0.2f * y;
            pk.o[v] = f2bf(y);
          }
          *(uint4*)(hout + (((size_t)ml * HH + h) * WW + px) * CC + cob) = pk.q;
        }
      }
    }
  } else {
    float mx[8];
    #pragma unroll
    for (int v = 0; v < 8; ++v) mx[v] = __uint_as_float(0xFF800000u);
    #pragma unroll
    for (int r = 0; r < 2; ++r) {
      #pragma unroll
      for (int j = 0; j < 4; ++j) {
        int px = w0 + j * 16 + nidx;
        if (px < WW) {
          #pragma unroll
          for (int v = 0; v < 8; ++v) mx[v] = fmaxf(mx[v], acc[r][j][v] + bias[cob + v]);
        }
      }
    }
    #pragma unroll
    for (int off = 1; off < 16; off <<= 1)
      #pragma unroll
      for (int v = 0; v < 8; ++v) mx[v] = fmaxf(mx[v], __shfl_xor(mx[v], off));
    if (nidx == 0) {
      int mg = map_base + ml;
      #pragma unroll
      for (int v = 0; v < 8; ++v) atomicMaxF(maxbuf + (size_t)mg * CC + cob + v, mx[v]);
    }
  }
}

// --------------------------------------------- per-row l2 norms -> feat
// rss[n][which][k] = squared norm of normalized row (for 2nd-stage norm)
__global__ __launch_bounds__(128)
void k_rownorm(const float* vlad, const float* maxbuf, float* feat, float* rss) {
  __shared__ float red[128];
  int which = blockIdx.x & 1, k = blockIdx.x >> 1, n = blockIdx.y, t = threadIdx.x;
  const float* src = (which == 0) ? vlad : maxbuf;
  float v = src[((size_t)n * KK + k) * CC + t];
  red[t] = v * v; __syncthreads();
  for (int st = 64; st > 0; st >>= 1) { if (t < st) red[t] += red[t + st]; __syncthreads(); }
  float ssq = red[0];
  float nr = fmaxf(sqrtf(ssq), 1e-12f);
  feat[(size_t)n * 2 * KK * CC + which * KK * CC + k * CC + t] = v / nr;
  if (t == 0) rss[(n * 2 + which) * KK + k] = ssq / (nr * nr);
}

// --------------------------------------------- MLP + final l2 norm
__global__ __launch_bounds__(256)
void k_mlp(const float* feat, const float* rss,
           const float* mlp_w, const float* mlp_b, float* out) {
  __shared__ float fs[2048];
  __shared__ float red[256];
  int n = blockIdx.x, t = threadIdx.x;
  float s0 = 0.f, s1 = 0.f;
  for (int k = 0; k < KK; ++k) { s0 += rss[(n * 2 + 0) * KK + k]; s1 += rss[(n * 2 + 1) * KK + k]; }
  float inv0 = 1.0f / fmaxf(sqrtf(s0), 1e-12f);
  float inv1 = 1.0f / fmaxf(sqrtf(s1), 1e-12f);
  const float* fb = feat + (size_t)n * 2 * KK * CC;
  float acc = 0.f;
  for (int big = 0; big < 2 * KK * CC; big += 2048) {
    __syncthreads();
    for (int i = t; i < 2048; i += 256) {
      float f = fb[big + i];
      fs[i] = f * ((big + i < KK * CC) ? inv0 : inv1);
    }
    __syncthreads();
    const float* wr = mlp_w + (size_t)t * (2 * KK * CC) + big;
    for (int i = 0; i < 2048; ++i) acc += fs[i] * wr[i];
  }
  acc += mlp_b[t];
  red[t] = acc * acc; __syncthreads();
  for (int st = 128; st > 0; st >>= 1) { if (t < st) red[t] += red[t + st]; __syncthreads(); }
  float nr = fmaxf(sqrtf(red[0]), 1e-12f);
  out[n * OUTD + t] = acc / nr;
}

// -----------------------------------------------------------------------
extern "C" void kernel_launch(void* const* d_in, const int* in_sizes, int n_in,
                              void* d_out, int out_size, void* d_ws, size_t ws_size,
                              hipStream_t stream) {
  const float* x      = (const float*)d_in[0];
  const float* cents  = (const float*)d_in[1];
  const float* conv_w = (const float*)d_in[2];
  const float* conv_b = (const float*)d_in[3];
  const float* mlp_w  = (const float*)d_in[4];
  const float* mlp_b  = (const float*)d_in[5];
  const float* c1_w   = (const float*)d_in[6];
  const float* c1_b   = (const float*)d_in[7];
  const float* bn1_g  = (const float*)d_in[8];
  const float* bn1_b  = (const float*)d_in[9];
  const float* bn1_m  = (const float*)d_in[10];
  const float* bn1_v  = (const float*)d_in[11];
  const float* c2_w   = (const float*)d_in[12];
  const float* c2_b   = (const float*)d_in[13];
  const float* bn2_g  = (const float*)d_in[14];
  const float* bn2_b  = (const float*)d_in[15];
  const float* bn2_m  = (const float*)d_in[16];
  const float* bn2_v  = (const float*)d_in[17];
  const float* c3_w   = (const float*)d_in[18];
  const float* c3_b   = (const float*)d_in[19];
  float* out = (float*)d_out;

  char* ws = (char*)d_ws;
  size_t off = 0;
  auto alloc = [&](size_t b) { size_t r = off; off += (b + 255) & ~(size_t)255; return r; };
  float*          soft    = (float*)         (ws + alloc((size_t)NN * KK * PP * 4));
  unsigned short* distb   = (unsigned short*)(ws + alloc((size_t)NN * KK * PP * 2));
  float*          softsum = (float*)         (ws + alloc((size_t)NN * KK * 4));
  float*          vlad    = (float*)         (ws + alloc((size_t)NN * KK * CC * 4));
  float*          maxbuf  = (float*)         (ws + alloc((size_t)NN * KK * CC * 4));
  float*          feat    = (float*)         (ws + alloc((size_t)NN * 2 * KK * CC * 4));
  float*          rss     = (float*)         (ws + alloc((size_t)NN * 2 * KK * 4));
  unsigned short* wpk2    = (unsigned short*)(ws + alloc((size_t)25 * CC * CC * 2));
  unsigned short* wpk3    = (unsigned short*)(ws + alloc((size_t)25 * CC * CC * 2));
  unsigned short* h1      = (unsigned short*)(ws + alloc((size_t)MCHUNK * PP * CC * 2));
  unsigned short* h2      = (unsigned short*)(ws + alloc((size_t)MCHUNK * PP * CC * 2));

  k_init<<<(NN * KK * CC + 255) / 256, 256, 0, stream>>>(maxbuf);
  int npack = 25 * CC * CC;
  k_pack<<<(npack + 255) / 256, 256, 0, stream>>>(c2_w, wpk2);
  k_pack<<<(npack + 255) / 256, 256, 0, stream>>>(c3_w, wpk3);

  size_t sh_assign = (size_t)(CC * 128 + 128 * 65) * 4;
  k_assign<<<dim3((PP + 127) / 128, NN), 128, sh_assign, stream>>>(
      x, conv_w, conv_b, cents, soft, distb);
  k_softsum<<<NN * KK, 256, 0, stream>>>(soft, softsum);
  size_t sh_vlad = (size_t)(128 * 129 + 128) * 4;
  k_vlad<<<dim3(KK, NN), 128, sh_vlad, stream>>>(x, soft, softsum, cents, vlad);

  size_t sh_conv = (size_t)(6 * 68 * 128 + 128 * 128) * 2;
  for (int mc = 0; mc < (NN * KK) / MCHUNK; ++mc) {
    int mb = mc * MCHUNK;
    k_conv1<<<dim3(HH, MCHUNK), 256, 0, stream>>>(
        distb, c1_w, c1_b, bn1_g, bn1_b, bn1_m, bn1_v, h1, mb);
    k_convg<<<dim3(4, HH / 2, MCHUNK), 256, sh_conv, stream>>>(
        h1, wpk2, c2_b, bn2_g, bn2_b, bn2_m, bn2_v, h2, maxbuf, 0, mb);
    k_convg<<<dim3(4, HH / 2, MCHUNK), 256, sh_conv, stream>>>(
        h2, wpk3, c3_b, bn2_g, bn2_b, bn2_m, bn2_v, h1 /*unused*/, maxbuf, 1, mb);
  }

  k_rownorm<<<dim3(2 * KK, NN), 128, 0, stream>>>(vlad, maxbuf, feat, rss);
  k_mlp<<<NN, 256, 0, stream>>>(feat, rss, mlp_w, mlp_b, out);
}
